// MambaFrameEncoder_14723147891396
// MI455X (gfx1250) — compile-verified, matrix operand order checked by reading, not run on hardware
//
#include <hip/hip_runtime.h>
#include <hip/hip_bf16.h>

// ---------------------------------------------------------------------------
// Mamba frame encoder for gfx1250 (MI455X).
// Shapes: B=4, H=40, W=120 -> L=4800, T=B*L=19200 tokens, V=256, DM=128,
// NL=2, NS=16, EXP=2 -> DI=256, DR=8, K(conv)=4.
// ---------------------------------------------------------------------------

#define Bz   4
#define Hh   40
#define Ww   120
#define Ll   (Hh * Ww)        // 4800
#define Tt   (Bz * Ll)        // 19200
#define DM   128
#define NL   2
#define NS   16
#define DI   256
#define DR   8
#define KC   4

typedef __attribute__((ext_vector_type(16))) _Float16 v16h;
typedef __attribute__((ext_vector_type(8)))  _Float16 v8h;
typedef __attribute__((ext_vector_type(8)))  float    v8f;

__device__ __forceinline__ float silu_f(float v) {
    return v / (1.0f + __expf(-v));
}
__device__ __forceinline__ float softplus_f(float v) {
    return (v > 20.0f) ? v : logf(1.0f + __expf(v));
}

// ---------------------------------------------------------------------------
// Embedding + positional gather: x[t, d] = emb[grid[t], d] + pos[t % L, d]
// ---------------------------------------------------------------------------
__global__ void embed_kernel(const int* __restrict__ grid,
                             const float* __restrict__ emb,
                             const float* __restrict__ pos,
                             float* __restrict__ x) {
    int g = blockIdx.x * blockDim.x + threadIdx.x;       // over T*DM
    if (g >= Tt * DM) return;
    int t = g / DM, d = g % DM;
    int tok = grid[t];
    x[g] = emb[tok * DM + d] + pos[(t % Ll) * DM + d];
}

// ---------------------------------------------------------------------------
// LayerNorm over DM=128 per token, emit f16 A-matrix for WMMA.
// ---------------------------------------------------------------------------
__global__ void ln_to_f16_kernel(const float* __restrict__ x,
                                 const float* __restrict__ w,
                                 const float* __restrict__ b,
                                 _Float16* __restrict__ out) {
    int t = blockIdx.x;
    int d = threadIdx.x;                                  // 128 threads
    float v = x[t * DM + d];
    __shared__ float s1[DM], s2[DM];
    s1[d] = v; s2[d] = v * v;
    __syncthreads();
    for (int o = DM / 2; o > 0; o >>= 1) {
        if (d < o) { s1[d] += s1[d + o]; s2[d] += s2[d + o]; }
        __syncthreads();
    }
    float m   = s1[0] * (1.0f / DM);
    float var = s2[0] * (1.0f / DM) - m * m;
    float r   = rsqrtf(var + 1e-5f);
    out[t * DM + d] = (_Float16)((v - m) * r * w[d] + b[d]);
}

// ---------------------------------------------------------------------------
// f32 -> f16 weight staging.
// ---------------------------------------------------------------------------
__global__ void cvt_f16_kernel(const float* __restrict__ src,
                               _Float16* __restrict__ dst, int n) {
    int g = blockIdx.x * blockDim.x + threadIdx.x;
    if (g < n) dst[g] = (_Float16)src[g];
}

// ---------------------------------------------------------------------------
// WMMA GEMM: C[M,N] = A[M,K] * W[N,K]^T (+ Cin residual).
// Each wave owns a 16x64 output strip: 1 M-tile x 4 N-tiles, so every
// K-step of 32 loads ONE A fragment and issues FOUR v_wmma_f32_16x16x32_f16
// (register-level A reuse; 4 WMMA per 10 b128 loads instead of 1 per 4).
// Per-lane layouts follow ISA 7.12.2:
//   A: lane L -> row m = L%16, koff = (L>>4)*8, halves {k..k+7, k+16..k+23}
//   B (= W row n, since B[k][n] = W[n][k]): mirrored layout with n = L%16
//   C/D: vgpr r -> row tm*16 + r + (L>>4)*8, col n0 + nt*16 + (L&15)
// ---------------------------------------------------------------------------
template <int K>
__global__ __launch_bounds__(128)
void wmma_gemm_kernel(const _Float16* __restrict__ A,
                      const _Float16* __restrict__ W,
                      const float* __restrict__ Cin,   // may be null
                      float* __restrict__ Cout,
                      int M, int N) {
    int wid  = (blockIdx.x * blockDim.x + threadIdx.x) >> 5;   // wave id
    int lane = threadIdx.x & 31;
    int stripsN = N >> 6;                                      // N/64
    int strips  = (M >> 4) * stripsN;
    if (wid >= strips) return;                                 // whole-wave
    int tm = wid / stripsN;
    int n0 = (wid % stripsN) * 64;

    int m    = tm * 16 + (lane & 15);
    int koff = (lane >> 4) * 8;
    const _Float16* Arow = A + (size_t)m * K + koff;
    const _Float16* Wrow = W + (size_t)(n0 + (lane & 15)) * K + koff;

    int cm = tm * 16 + (lane >> 4) * 8;
    int cn = n0 + (lane & 15);

    v8f c0 = {}, c1 = {}, c2 = {}, c3 = {};
    if (Cin) {
#pragma unroll
        for (int r = 0; r < 8; ++r) {
            const float* row = Cin + (size_t)(cm + r) * N + cn;
            c0[r] = row[0];
            c1[r] = row[16];
            c2[r] = row[32];
            c3[r] = row[48];
        }
    }

#pragma unroll
    for (int kb = 0; kb < K; kb += 32) {
        v8h alo = *(const v8h*)(Arow + kb);
        v8h ahi = *(const v8h*)(Arow + kb + 16);
        v16h a;
#pragma unroll
        for (int j = 0; j < 8; ++j) { a[j] = alo[j]; a[8 + j] = ahi[j]; }

        v8f* accs[4] = { &c0, &c1, &c2, &c3 };
#pragma unroll
        for (int nt = 0; nt < 4; ++nt) {
            const _Float16* Wp = Wrow + (size_t)nt * 16 * K + kb;
            v8h blo = *(const v8h*)(Wp);
            v8h bhi = *(const v8h*)(Wp + 16);
            v16h bm;
#pragma unroll
            for (int j = 0; j < 8; ++j) { bm[j] = blo[j]; bm[8 + j] = bhi[j]; }
            *accs[nt] = __builtin_amdgcn_wmma_f32_16x16x32_f16(
                /*neg_a=*/false, a, /*neg_b=*/false, bm,
                /*c_mod=*/(short)0, *accs[nt],
                /*reuse_a=*/false, /*reuse_b=*/false);
        }
    }

#pragma unroll
    for (int r = 0; r < 8; ++r) {
        float* row = Cout + (size_t)(cm + r) * N + cn;
        row[0]  = c0[r];
        row[16] = c1[r];
        row[32] = c2[r];
        row[48] = c3[r];
    }
}

// ---------------------------------------------------------------------------
// Depthwise causal conv (K=4) + bias + SiLU.  xin = xz[..., :DI].
// ---------------------------------------------------------------------------
__global__ void conv_silu_kernel(const float* __restrict__ xz,
                                 const float* __restrict__ cw,  // (DI, K)
                                 const float* __restrict__ cb,  // (DI,)
                                 float* __restrict__ xc) {
    int g = blockIdx.x * blockDim.x + threadIdx.x;       // over T*DI
    if (g >= Tt * DI) return;
    int t = g / DI, d = g % DI;
    int b = t / Ll, l = t % Ll;
    float s = cb[d];
#pragma unroll
    for (int j = 0; j < KC; ++j) {
        int li = l - (KC - 1) + j;
        if (li >= 0)
            s += xz[(size_t)(b * Ll + li) * (2 * DI) + d] * cw[d * KC + j];
    }
    xc[g] = silu_f(s);
}

// ---------------------------------------------------------------------------
// x_proj: dbl[t, e] = sum_d xc[t, d] * xpw[e, d],  e in [0, DR+2*NS=40).
// ---------------------------------------------------------------------------
__global__ void xproj_kernel(const float* __restrict__ xc,
                             const float* __restrict__ xpw, // (40, DI)
                             float* __restrict__ dbl) {
    int g = blockIdx.x * blockDim.x + threadIdx.x;       // over T*40
    if (g >= Tt * (DR + 2 * NS)) return;
    int t = g / (DR + 2 * NS), e = g % (DR + 2 * NS);
    const float* xr = xc + (size_t)t * DI;
    const float* wr = xpw + (size_t)e * DI;
    float s = 0.0f;
#pragma unroll 8
    for (int d = 0; d < DI; ++d) s += xr[d] * wr[d];
    dbl[g] = s;
}

// ---------------------------------------------------------------------------
// dt_proj + softplus: delta[t, d] = softplus(sum_r dt[t,r]*dpw[d,r] + dpb[d])
// ---------------------------------------------------------------------------
__global__ void dtproj_kernel(const float* __restrict__ dbl,
                              const float* __restrict__ dpw, // (DI, DR)
                              const float* __restrict__ dpb, // (DI,)
                              float* __restrict__ delta) {
    int g = blockIdx.x * blockDim.x + threadIdx.x;       // over T*DI
    if (g >= Tt * DI) return;
    int t = g / DI, d = g % DI;
    float s = dpb[d];
#pragma unroll
    for (int r = 0; r < DR; ++r)
        s += dbl[(size_t)t * (DR + 2 * NS) + r] * dpw[d * DR + r];
    delta[g] = softplus_f(s);
}

// ---------------------------------------------------------------------------
// Selective scan + D skip + SiLU(z) gating, fused; writes f16 y for out_proj.
// 16 lanes per (b,d) channel, one SSM state n per lane, sequential over L.
// Cross-lane Σ h·C via 4-step xor-shuffle tree within the 16-lane group.
// ---------------------------------------------------------------------------
__global__ void ssm_scan_kernel(const float* __restrict__ delta,
                                const float* __restrict__ dbl,
                                const float* __restrict__ xc,
                                const float* __restrict__ xz,   // z at [DI..2DI)
                                const float* __restrict__ Alog, // (DI, NS)
                                const float* __restrict__ Dp,   // (DI,)
                                _Float16* __restrict__ yh) {
    int g = blockIdx.x * blockDim.x + threadIdx.x;       // B*DI*NS = 16384
    if (g >= Bz * DI * NS) return;
    int n = g & (NS - 1);
    int d = (g >> 4) & (DI - 1);
    int b = g >> 12;

    float eA = __expf(Alog[d * NS + n]);
    float Dd = Dp[d];
    float h = 0.0f;

    for (int l = 0; l < Ll; ++l) {
        size_t t = (size_t)b * Ll + l;
        float dv = delta[t * DI + d];        // broadcast across 16 lanes
        float xv = xc[t * DI + d];
        float Bv = dbl[t * (DR + 2 * NS) + DR + n];
        float Cv = dbl[t * (DR + 2 * NS) + DR + NS + n];
        float a  = __expf(-dv * eA);
        h = a * h + dv * Bv * xv;
        float p = h * Cv;
        p += __shfl_xor(p, 8, 16);
        p += __shfl_xor(p, 4, 16);
        p += __shfl_xor(p, 2, 16);
        p += __shfl_xor(p, 1, 16);
        if (n == 0) {
            float zv = xz[t * (2 * DI) + DI + d];
            yh[t * DI + d] = (_Float16)((p + xv * Dd) * silu_f(zv));
        }
    }
}

// ---------------------------------------------------------------------------
// Final: zero d_out, then per-token LayerNorm + atomic mean over L.
// ---------------------------------------------------------------------------
__global__ void zero_out_kernel(float* __restrict__ out, int n) {
    int g = blockIdx.x * blockDim.x + threadIdx.x;
    if (g < n) out[g] = 0.0f;
}

__global__ void final_reduce_kernel(const float* __restrict__ x,
                                    const float* __restrict__ fw,
                                    const float* __restrict__ fb,
                                    float* __restrict__ out) {
    int t = blockIdx.x;
    int d = threadIdx.x;                                  // 128 threads
    int b = t / Ll;
    float v = x[t * DM + d];
    __shared__ float s1[DM], s2[DM];
    s1[d] = v; s2[d] = v * v;
    __syncthreads();
    for (int o = DM / 2; o > 0; o >>= 1) {
        if (d < o) { s1[d] += s1[d + o]; s2[d] += s2[d + o]; }
        __syncthreads();
    }
    float m   = s1[0] * (1.0f / DM);
    float var = s2[0] * (1.0f / DM) - m * m;
    float r   = rsqrtf(var + 1e-5f);
    float y   = (v - m) * r * fw[d] + fb[d];
    atomicAdd(&out[b * DM + d], y * (1.0f / Ll));
}

// ---------------------------------------------------------------------------
// Host-side launcher.
// ---------------------------------------------------------------------------
extern "C" void kernel_launch(void* const* d_in, const int* in_sizes, int n_in,
                              void* d_out, int out_size, void* d_ws, size_t ws_size,
                              hipStream_t stream) {
    (void)in_sizes; (void)n_in; (void)out_size; (void)ws_size;

    const int*   grid_i = (const int*)  d_in[0];
    const float* emb    = (const float*)d_in[1];
    const float* pos    = (const float*)d_in[2];
    const float* norm_w = (const float*)d_in[3];
    const float* norm_b = (const float*)d_in[4];
    const float* ipw    = (const float*)d_in[5];   // (NL, 512, 128)
    const float* cw     = (const float*)d_in[6];   // (NL, DI, 1, KC)
    const float* cb     = (const float*)d_in[7];   // (NL, DI)
    const float* xpw    = (const float*)d_in[8];   // (NL, 40, DI)
    const float* dpw    = (const float*)d_in[9];   // (NL, DI, DR)
    const float* dpb    = (const float*)d_in[10];  // (NL, DI)
    const float* Alog   = (const float*)d_in[11];  // (NL, DI, NS)
    const float* Dp     = (const float*)d_in[12];  // (NL, DI)
    const float* opw    = (const float*)d_in[13];  // (NL, DM, DI)
    const float* fw     = (const float*)d_in[14];
    const float* fb     = (const float*)d_in[15];
    float* out = (float*)d_out;

    // Workspace layout (256B aligned).
    char* ws = (char*)d_ws;
    size_t off = 0;
    auto alloc = [&](size_t bytes) -> char* {
        char* p = ws + off;
        off += (bytes + 255) & ~(size_t)255;
        return p;
    };
    float*    x     = (float*)   alloc((size_t)Tt * DM * 4);       // activations
    _Float16* xnh   = (_Float16*)alloc((size_t)Tt * DM * 2);       // LN(x) f16
    float*    xz    = (float*)   alloc((size_t)Tt * 2 * DI * 4);   // in_proj out
    float*    xc    = (float*)   alloc((size_t)Tt * DI * 4);       // conv+silu
    float*    dbl   = (float*)   alloc((size_t)Tt * (DR + 2 * NS) * 4);
    float*    delta = (float*)   alloc((size_t)Tt * DI * 4);
    _Float16* yh    = (_Float16*)alloc((size_t)Tt * DI * 2);       // gated y f16
    _Float16* ipw_h = (_Float16*)alloc((size_t)NL * 2 * DI * DM * 2);
    _Float16* opw_h = (_Float16*)alloc((size_t)NL * DM * DI * 2);

    // 0) Stage all weights in f16 once per call.
    cvt_f16_kernel<<<(NL * 2 * DI * DM + 255) / 256, 256, 0, stream>>>(
        ipw, ipw_h, NL * 2 * DI * DM);
    cvt_f16_kernel<<<(NL * DM * DI + 255) / 256, 256, 0, stream>>>(
        opw, opw_h, NL * DM * DI);

    // 1) Embedding + positional.
    embed_kernel<<<(Tt * DM + 255) / 256, 256, 0, stream>>>(grid_i, emb, pos, x);

    for (int i = 0; i < NL; ++i) {
        const float* nw_i  = norm_w + i * DM;
        const float* nb_i  = norm_b + i * DM;
        const float* cw_i  = cw  + (size_t)i * DI * KC;
        const float* cb_i  = cb  + (size_t)i * DI;
        const float* xpw_i = xpw + (size_t)i * (DR + 2 * NS) * DI;
        const float* dpw_i = dpw + (size_t)i * DI * DR;
        const float* dpb_i = dpb + (size_t)i * DI;
        const float* Al_i  = Alog + (size_t)i * DI * NS;
        const float* Dp_i  = Dp  + (size_t)i * DI;
        const _Float16* ipwh_i = ipw_h + (size_t)i * 2 * DI * DM;
        const _Float16* opwh_i = opw_h + (size_t)i * DM * DI;

        // 2) LayerNorm -> f16 A-matrix.
        ln_to_f16_kernel<<<Tt, DM, 0, stream>>>(x, nw_i, nb_i, xnh);

        // 3) in_proj WMMA GEMM: (19200 x 128) x (512 x 128)^T -> xz.
        {
            int strips = (Tt / 16) * (2 * DI / 64);          // 9600 waves
            wmma_gemm_kernel<DM><<<(strips + 3) / 4, 128, 0, stream>>>(
                xnh, ipwh_i, nullptr, xz, Tt, 2 * DI);
        }

        // 4) Depthwise causal conv + SiLU.
        conv_silu_kernel<<<(Tt * DI + 255) / 256, 256, 0, stream>>>(xz, cw_i, cb_i, xc);

        // 5) x_proj (N=40, VALU).
        xproj_kernel<<<(Tt * (DR + 2 * NS) + 255) / 256, 256, 0, stream>>>(xc, xpw_i, dbl);

        // 6) dt_proj + softplus.
        dtproj_kernel<<<(Tt * DI + 255) / 256, 256, 0, stream>>>(dbl, dpw_i, dpb_i, delta);

        // 7) Selective scan, fused with D-skip and SiLU(z) gating -> yh (f16).
        ssm_scan_kernel<<<(Bz * DI * NS + 255) / 256, 256, 0, stream>>>(
            delta, dbl, xc, xz, Al_i, Dp_i, yh);

        // 8) out_proj WMMA GEMM with residual accumulate into x.
        {
            int strips = (Tt / 16) * (DM / 64);              // 2400 waves
            wmma_gemm_kernel<DI><<<(strips + 3) / 4, 128, 0, stream>>>(
                yh, opwh_i, x, x, Tt, DM);
        }
    }

    // 9) Final LayerNorm + mean over L.
    zero_out_kernel<<<1, 512, 0, stream>>>(out, Bz * DM);
    final_reduce_kernel<<<Tt, DM, 0, stream>>>(x, fw, fb, out);
}
